// CrossAttention_70385924047043
// MI455X (gfx1250) — matrix operand order repaired and checked by reading, not compiled, not run on hardware
//
#include <hip/hip_runtime.h>

// ---------------------------------------------------------------------------
// CDNA5 (gfx1250) cross-attention:  out = softmax(Q K^T) V  with
//   Q = query@Wq+bq, K = key@Wk+bk, V = value@Wv+bv   (no 1/sqrt scaling)
// All GEMMs use v_wmma_f32_16x16x32_f16 (f32 accumulate).
// Double-buffered LDS B-tiles (one barrier / k-step, global latency hidden
// under WMMA), b128-vectorized global traffic, batched B-fragment LDS reads,
// global_prefetch_b8 two tiles ahead.
// ---------------------------------------------------------------------------

typedef __attribute__((ext_vector_type(16))) _Float16 v16h;
typedef __attribute__((ext_vector_type(8)))  _Float16 v8h;
typedef __attribute__((ext_vector_type(8)))  float    v8f;

#define WMMA_F16F32(a, b, c)                                                   \
  __builtin_amdgcn_wmma_f32_16x16x32_f16(false, (a), false, (b), (short)0,     \
                                         (c), false, false)

// LDS B-tile: [64 cols][48 halves] ; col*96B is 32B-aligned, kb*2B in {0,32}
// => every v16h fragment read is 32B-aligned.
#define BT_STRIDE 48

// A-fragment (16x32 f16) from a row-major f16 row; contiguous layout:
// lane-half kbA in {0,8}; k = 2*kbA + e.
__device__ __forceinline__ v16h load_a_row_f16(const _Float16* __restrict__ row,
                                               int kk, int kbA) {
  const v8h* p = (const v8h*)(row + kk + 2 * kbA);
  v8h lo = p[0];   // k = kk+2*kbA .. +7
  v8h hi = p[1];   // k = kk+2*kbA+8 .. +15
  v16h a;
#pragma unroll
  for (int e = 0; e < 8; ++e) { a[e] = lo[e]; a[e + 8] = hi[e]; }
  return a;
}

// A-fragment from a row-major f32 row, converted to f16.
__device__ __forceinline__ v16h load_a_row_f32(const float* __restrict__ row,
                                               int kk, int kbA) {
  const float4* p0 = (const float4*)(row + kk + 2 * kbA);
  const float4* p1 = (const float4*)(row + kk + 2 * kbA + 8);
  float4 a0 = p0[0], a1 = p0[1];
  float4 a2 = p1[0], a3 = p1[1];
  v16h a;
  a[0]  = (_Float16)a0.x; a[1]  = (_Float16)a0.y;
  a[2]  = (_Float16)a0.z; a[3]  = (_Float16)a0.w;
  a[4]  = (_Float16)a1.x; a[5]  = (_Float16)a1.y;
  a[6]  = (_Float16)a1.z; a[7]  = (_Float16)a1.w;
  a[8]  = (_Float16)a2.x; a[9]  = (_Float16)a2.y;
  a[10] = (_Float16)a2.z; a[11] = (_Float16)a2.w;
  a[12] = (_Float16)a3.x; a[13] = (_Float16)a3.y;
  a[14] = (_Float16)a3.z; a[15] = (_Float16)a3.w;
  return a;
}

// ---------------------------------------------------------------------------
// Projection GEMM: out[M,N] (f16) = A[M,D] (f32) @ W[D,N] (f32) + bias[N]
// 256 threads = 8 waves; block tile 128(M) x 64(N); wave tile 16 x 64.
// W tile (32x64) double-buffered in LDS, stored transposed.
// ---------------------------------------------------------------------------
__global__ __launch_bounds__(256)
void proj_kernel(const float* __restrict__ A, const float* __restrict__ W,
                 const float* __restrict__ bias, _Float16* __restrict__ out,
                 int D, int N) {
  __shared__ _Float16 bt0[64 * BT_STRIDE];   // 6 KB
  __shared__ _Float16 bt1[64 * BT_STRIDE];   // 6 KB

  const int tid  = threadIdx.x;
  const int lane = tid & 31;
  const int wave = tid >> 5;
  const int rowBase = blockIdx.y * 128 + wave * 16;
  const int colBase = blockIdx.x * 64;
  const int r   = lane & 15;
  const int hi  = lane >> 4;   // 0 or 1
  const int kbA = hi * 8;
  const int kbB = hi * 16;

  // Staging coords: each of 256 threads handles 8 f32 of the 32x64 W tile.
  const int kRow = tid >> 3;          // 0..31
  const int nOff = (tid & 7) * 8;     // 0,8,...,56

  const float* __restrict__ arow = A + (size_t)(rowBase + r) * D;

  v8f acc[4] = {};

  auto stage_load = [&](int kk, float4& w0, float4& w1) {
    const float4* gp =
        (const float4*)(W + (size_t)(kk + kRow) * N + colBase + nOff);
    w0 = gp[0];
    w1 = gp[1];
    if (kk + 64 < D) {  // gfx1250 global_prefetch_b8, two tiles ahead
      __builtin_prefetch(W + (size_t)(kk + 64 + kRow) * N + colBase + nOff,
                         0, 1);
    }
  };
  auto lds_store = [&](_Float16* buf, float4 w0, float4 w1) {
    buf[(nOff + 0) * BT_STRIDE + kRow] = (_Float16)w0.x;
    buf[(nOff + 1) * BT_STRIDE + kRow] = (_Float16)w0.y;
    buf[(nOff + 2) * BT_STRIDE + kRow] = (_Float16)w0.z;
    buf[(nOff + 3) * BT_STRIDE + kRow] = (_Float16)w0.w;
    buf[(nOff + 4) * BT_STRIDE + kRow] = (_Float16)w1.x;
    buf[(nOff + 5) * BT_STRIDE + kRow] = (_Float16)w1.y;
    buf[(nOff + 6) * BT_STRIDE + kRow] = (_Float16)w1.z;
    buf[(nOff + 7) * BT_STRIDE + kRow] = (_Float16)w1.w;
  };
  auto compute = [&](const _Float16* buf, int kk) {
    v16h a  = load_a_row_f32(arow, kk, kbA);
    v16h b0 = *(const v16h*)(&buf[(0 * 16 + r) * BT_STRIDE + kbB]);
    v16h b1 = *(const v16h*)(&buf[(1 * 16 + r) * BT_STRIDE + kbB]);
    v16h b2 = *(const v16h*)(&buf[(2 * 16 + r) * BT_STRIDE + kbB]);
    v16h b3 = *(const v16h*)(&buf[(3 * 16 + r) * BT_STRIDE + kbB]);
    acc[0] = WMMA_F16F32(a, b0, acc[0]);
    acc[1] = WMMA_F16F32(a, b1, acc[1]);
    acc[2] = WMMA_F16F32(a, b2, acc[2]);
    acc[3] = WMMA_F16F32(a, b3, acc[3]);
  };

  float4 g0, g1;
  stage_load(0, g0, g1);
  lds_store(bt0, g0, g1);
  __syncthreads();

  // D is a multiple of 64 (1024 or 768).
  for (int kk = 0; kk < D; kk += 64) {
    const bool more = (kk + 64) < D;
    float4 h0, h1;
    stage_load(kk + 32, h0, h1);   // next tile's global loads in flight
    compute(bt0, kk);              // WMMA from current buffer
    lds_store(bt1, h0, h1);
    __syncthreads();

    if (more) stage_load(kk + 64, g0, g1);
    compute(bt1, kk + 32);
    if (more) lds_store(bt0, g0, g1);
    __syncthreads();
  }

#pragma unroll
  for (int s = 0; s < 4; ++s) {
    const int col = colBase + s * 16 + r;
    const float bv = bias[col];
#pragma unroll
    for (int e = 0; e < 8; ++e) {
      const int row = rowBase + hi * 8 + e;
      out[(size_t)row * N + col] = (_Float16)(acc[s][e] + bv);
    }
  }
}

// ---------------------------------------------------------------------------
// Scores: S[LQ,LK] (f32) = Q[LQ,H] (f16) @ K[LK,H]^T (f16)
// K tile (64 keys x 32 h) double-buffered in LDS.
// ---------------------------------------------------------------------------
__global__ __launch_bounds__(256)
void scores_kernel(const _Float16* __restrict__ Q, const _Float16* __restrict__ K,
                   float* __restrict__ S, int LKc, int Hc) {
  __shared__ _Float16 kt0[64 * BT_STRIDE];
  __shared__ _Float16 kt1[64 * BT_STRIDE];

  const int tid  = threadIdx.x;
  const int lane = tid & 31;
  const int wave = tid >> 5;
  const int rowBase = blockIdx.y * 128 + wave * 16;
  const int colBase = blockIdx.x * 64;
  const int r   = lane & 15;
  const int hi  = lane >> 4;
  const int kbA = hi * 8;
  const int kbB = hi * 16;

  // Staging coords: 256 threads x 8 halves = 64 keys x 32 h.
  const int keyRow = tid >> 2;        // 0..63
  const int hOff   = (tid & 3) * 8;   // 0,8,16,24

  const _Float16* __restrict__ qrow = Q + (size_t)(rowBase + r) * Hc;
  const _Float16* __restrict__ krow = K + (size_t)(colBase + keyRow) * Hc;

  v8f acc[4] = {};

  auto stage_load = [&](int kk, v8h& g) {
    g = *(const v8h*)(krow + kk + hOff);
    if (kk + 64 < Hc) {
      __builtin_prefetch(krow + kk + 64 + hOff, 0, 1);
    }
  };
  auto lds_store = [&](_Float16* buf, v8h g) {
    *(v8h*)(&buf[keyRow * BT_STRIDE + hOff]) = g;
  };
  auto compute = [&](const _Float16* buf, int kk) {
    v16h a  = load_a_row_f16(qrow, kk, kbA);
    v16h b0 = *(const v16h*)(&buf[(0 * 16 + r) * BT_STRIDE + kbB]);
    v16h b1 = *(const v16h*)(&buf[(1 * 16 + r) * BT_STRIDE + kbB]);
    v16h b2 = *(const v16h*)(&buf[(2 * 16 + r) * BT_STRIDE + kbB]);
    v16h b3 = *(const v16h*)(&buf[(3 * 16 + r) * BT_STRIDE + kbB]);
    acc[0] = WMMA_F16F32(a, b0, acc[0]);
    acc[1] = WMMA_F16F32(a, b1, acc[1]);
    acc[2] = WMMA_F16F32(a, b2, acc[2]);
    acc[3] = WMMA_F16F32(a, b3, acc[3]);
  };

  v8h g;
  stage_load(0, g);
  lds_store(kt0, g);
  __syncthreads();

  // Hc is a multiple of 64 (1024).
  for (int kk = 0; kk < Hc; kk += 64) {
    const bool more = (kk + 64) < Hc;
    v8h h;
    stage_load(kk + 32, h);
    compute(kt0, kk);
    lds_store(kt1, h);
    __syncthreads();

    if (more) stage_load(kk + 64, g);
    compute(kt1, kk + 32);
    if (more) lds_store(kt0, g);
    __syncthreads();
  }

#pragma unroll
  for (int s = 0; s < 4; ++s) {
    const int col = colBase + s * 16 + r;
#pragma unroll
    for (int e = 0; e < 8; ++e) {
      const int row = rowBase + hi * 8 + e;
      S[(size_t)row * LKc + col] = acc[s][e];
    }
  }
}

// ---------------------------------------------------------------------------
// Row softmax: P[row,:] (f16) = softmax(S[row,:]) ; one block per row.
// ---------------------------------------------------------------------------
__global__ __launch_bounds__(256)
void softmax_kernel(const float* __restrict__ S, _Float16* __restrict__ P,
                    int LKc) {
  __shared__ float buf[2048];
  __shared__ float red[256];
  const int row = blockIdx.x;
  const int tid = threadIdx.x;
  const float* __restrict__ srow = S + (size_t)row * LKc;

  float m = -3.4e38f;
  for (int i = tid; i < LKc; i += 256) {
    float x = srow[i];
    buf[i] = x;
    m = fmaxf(m, x);
  }
  red[tid] = m;
  __syncthreads();
#pragma unroll
  for (int off = 128; off > 0; off >>= 1) {
    if (tid < off) red[tid] = fmaxf(red[tid], red[tid + off]);
    __syncthreads();
  }
  m = red[0];
  __syncthreads();

  float sum = 0.0f;
  for (int i = tid; i < LKc; i += 256) {
    float e = __expf(buf[i] - m);
    buf[i] = e;
    sum += e;
  }
  red[tid] = sum;
  __syncthreads();
#pragma unroll
  for (int off = 128; off > 0; off >>= 1) {
    if (tid < off) red[tid] += red[tid + off];
    __syncthreads();
  }
  const float inv = 1.0f / red[0];

  for (int i = tid; i < LKc; i += 256) {
    P[(size_t)row * LKc + i] = (_Float16)(buf[i] * inv);
  }
}

// ---------------------------------------------------------------------------
// Output GEMM: O[LQ,H] (f32) = P[LQ,LK] (f16) @ V[LK,H] (f16)
// V tile (32 k x 64 n) double-buffered in LDS, stored transposed.
// ---------------------------------------------------------------------------
__global__ __launch_bounds__(256)
void out_kernel(const _Float16* __restrict__ P, const _Float16* __restrict__ V,
                float* __restrict__ O, int LKc, int Hc) {
  __shared__ _Float16 vt0[64 * BT_STRIDE];
  __shared__ _Float16 vt1[64 * BT_STRIDE];

  const int tid  = threadIdx.x;
  const int lane = tid & 31;
  const int wave = tid >> 5;
  const int rowBase = blockIdx.y * 128 + wave * 16;
  const int colBase = blockIdx.x * 64;
  const int r   = lane & 15;
  const int hi  = lane >> 4;
  const int kbA = hi * 8;
  const int kbB = hi * 16;

  // Staging coords: 256 threads x 8 halves = 32 k x 64 n (V rows coalesced).
  const int kRow = tid >> 3;          // 0..31
  const int nOff = (tid & 7) * 8;     // 0,8,...,56

  const _Float16* __restrict__ prow = P + (size_t)(rowBase + r) * LKc;

  v8f acc[4] = {};

  auto stage_load = [&](int kk, v8h& g) {
    g = *(const v8h*)(V + (size_t)(kk + kRow) * Hc + colBase + nOff);
    if (kk + 64 < LKc) {
      __builtin_prefetch(V + (size_t)(kk + 64 + kRow) * Hc + colBase + nOff,
                         0, 1);
    }
  };
  auto lds_store = [&](_Float16* buf, v8h g) {
#pragma unroll
    for (int i = 0; i < 8; ++i) {
      buf[(nOff + i) * BT_STRIDE + kRow] = g[i];
    }
  };
  auto compute = [&](const _Float16* buf, int kk) {
    v16h a  = load_a_row_f16(prow, kk, kbA);
    v16h b0 = *(const v16h*)(&buf[(0 * 16 + r) * BT_STRIDE + kbB]);
    v16h b1 = *(const v16h*)(&buf[(1 * 16 + r) * BT_STRIDE + kbB]);
    v16h b2 = *(const v16h*)(&buf[(2 * 16 + r) * BT_STRIDE + kbB]);
    v16h b3 = *(const v16h*)(&buf[(3 * 16 + r) * BT_STRIDE + kbB]);
    acc[0] = WMMA_F16F32(a, b0, acc[0]);
    acc[1] = WMMA_F16F32(a, b1, acc[1]);
    acc[2] = WMMA_F16F32(a, b2, acc[2]);
    acc[3] = WMMA_F16F32(a, b3, acc[3]);
  };

  v8h g;
  stage_load(0, g);
  lds_store(vt0, g);
  __syncthreads();

  // LKc is a multiple of 64 (2048).
  for (int kk = 0; kk < LKc; kk += 64) {
    const bool more = (kk + 64) < LKc;
    v8h h;
    stage_load(kk + 32, h);
    compute(vt0, kk);
    lds_store(vt1, h);
    __syncthreads();

    if (more) stage_load(kk + 64, g);
    compute(vt1, kk + 32);
    if (more) lds_store(vt0, g);
    __syncthreads();
  }

#pragma unroll
  for (int s = 0; s < 4; ++s) {
    const int col = colBase + s * 16 + r;
#pragma unroll
    for (int e = 0; e < 8; ++e) {
      const int row = rowBase + hi * 8 + e;
      O[(size_t)row * Hc + col] = acc[s][e];
    }
  }
}

// ---------------------------------------------------------------------------
// Host launcher
// ---------------------------------------------------------------------------
extern "C" void kernel_launch(void* const* d_in, const int* in_sizes, int n_in,
                              void* d_out, int out_size, void* d_ws,
                              size_t ws_size, hipStream_t stream) {
  (void)in_sizes; (void)n_in; (void)out_size; (void)ws_size;

  constexpr int Bn = 4, LQ = 2048, LK = 2048, D1 = 1024, D2 = 768, H = 1024;

  const float* query = (const float*)d_in[0];
  const float* key   = (const float*)d_in[1];
  const float* value = (const float*)d_in[2];
  const float* Wq    = (const float*)d_in[3];
  const float* bq    = (const float*)d_in[4];
  const float* Wk    = (const float*)d_in[5];
  const float* bk    = (const float*)d_in[6];
  const float* Wv    = (const float*)d_in[7];
  const float* bv    = (const float*)d_in[8];
  float*       out   = (float*)d_out;

  // Workspace layout (~75.5 MB):
  //   Qh: B*LQ*H f16 | Kh: B*LK*H f16 | Vh: B*LK*H f16
  //   Ph: LQ*LK f16 (per-batch) | Sf: LQ*LK f32 (per-batch)
  _Float16* Qh = (_Float16*)d_ws;
  _Float16* Kh = Qh + (size_t)Bn * LQ * H;
  _Float16* Vh = Kh + (size_t)Bn * LK * H;
  _Float16* Ph = Vh + (size_t)Bn * LK * H;
  float*    Sf = (float*)(Ph + (size_t)LQ * LK);

  dim3 blk(256);

  // Projections (flat over batch: [B*L, D] @ [D, H] + bias)
  proj_kernel<<<dim3(H / 64, (Bn * LQ) / 128), blk, 0, stream>>>(
      query, Wq, bq, Qh, D1, H);
  proj_kernel<<<dim3(H / 64, (Bn * LK) / 128), blk, 0, stream>>>(
      key, Wk, bk, Kh, D2, H);
  proj_kernel<<<dim3(H / 64, (Bn * LK) / 128), blk, 0, stream>>>(
      value, Wv, bv, Vh, D2, H);

  // Attention per batch (keeps S/P workspace per-batch sized)
  for (int b = 0; b < Bn; ++b) {
    const _Float16* Qb = Qh + (size_t)b * LQ * H;
    const _Float16* Kb = Kh + (size_t)b * LK * H;
    const _Float16* Vb = Vh + (size_t)b * LK * H;
    float* Ob = out + (size_t)b * LQ * H;

    scores_kernel<<<dim3(LK / 64, LQ / 128), blk, 0, stream>>>(Qb, Kb, Sf, LK, H);
    softmax_kernel<<<dim3(LQ), blk, 0, stream>>>(Sf, Ph, LK);
    out_kernel<<<dim3(H / 64, LQ / 128), blk, 0, stream>>>(Ph, Vb, Ob, LK, H);
  }
}
